// HAN_61306363183631
// MI455X (gfx1250) — matrix operand order repaired and checked by reading, not compiled, not run on hardware
//
#include <hip/hip_runtime.h>
#include <hip/hip_bf16.h>

// ---------------------------------------------------------------------------
// CDNA5 (gfx1250) HAN implementation: WMMA f16 GEMMs + L2-resident edge phase
// ---------------------------------------------------------------------------

typedef __attribute__((ext_vector_type(16))) _Float16 v16h;
typedef __attribute__((ext_vector_type(4)))  _Float16 v4h;
typedef __attribute__((ext_vector_type(8)))  float    v8f;

union V16H { v16h v; _Float16 h[16]; };

#define TBM 128
#define TBN 64
#define TKC 64
#define KPITCH (TKC + 4)   // half-pitch; 68*2B=136B rows keep 8B alignment

// Epilogue modes
#define EP_NONE 0
#define EP_RELU 1
#define EP_TANH_COLSUM 2

__device__ __forceinline__ unsigned pack2h(float x, float y) {
    auto p = __builtin_amdgcn_cvt_pkrtz(x, y);   // v_cvt_pk_rtz_f16_f32
    return __builtin_bit_cast(unsigned, p);
}

// ---------------------------------------------------------------------------
// WMMA GEMM: C[M,N] = A[M,K] @ B[K,N] + bias, f32 in/out, f16 multiply,
// f32 accumulate in registers (full K kept in accumulators).
// A staged [row][k] (row-major); B staged TRANSPOSED [col][k] so that each
// lane's WMMA B fragment is two contiguous 8-half runs -> vector ds_load_b64s.
// ep==EP_TANH_COLSUM: do not store C; accumulate sum_n tanh(C[n,c]) into
// colsum[c] (LDS reduce, then one global atomic per column per block).
// ---------------------------------------------------------------------------
__global__ __launch_bounds__(256) void wmma_gemm_k(
    const float* __restrict__ A, const float* __restrict__ B,
    const float* __restrict__ bias, float* __restrict__ C,
    int M, int N, int K, int ep, float* __restrict__ colsum)
{
    __shared__ _Float16 Ash[TBM][KPITCH];
    __shared__ _Float16 Bsh[TBN][KPITCH];
    __shared__ float    csred[TBN];

    const int tid  = threadIdx.x;
    const int wv   = tid >> 5;        // 8 waves
    const int lane = tid & 31;
    const int half = lane >> 4;
    const int r16  = lane & 15;
    const int m0   = blockIdx.x * TBM;
    const int n0   = blockIdx.y * TBN;

    if (tid < TBN) csred[tid] = 0.f;

    v8f acc[4] = {};

    for (int ks = 0; ks < K; ks += TKC) {
        __syncthreads();
        // Stage A tile: TBM x TKC f32 -> f16; one packed b32 store per float2.
        #pragma unroll
        for (int it = 0; it < 16; ++it) {
            int slot = it * 256 + tid;
            int row  = slot >> 5;          // TKC/2 = 32 float2 per row
            int c2   = slot & 31;
            float2 v = make_float2(0.f, 0.f);
            int gr = m0 + row;
            if (gr < M)
                v = *(const float2*)(A + (size_t)gr * K + ks + c2 * 2);
            *(unsigned*)&Ash[row][c2 * 2] = pack2h(v.x, v.y);
        }
        // Stage B tile transposed: Bsh[col][k]. Each thread loads two
        // K-adjacent elements of one column (coalesced along N), packs to
        // one dword, single ds_store_b32.
        #pragma unroll
        for (int it = 0; it < 8; ++it) {
            int slot = it * 256 + tid;
            int col  = slot & 63;
            int kp   = slot >> 6;          // 0..31 (pairs of k)
            const float* bp = B + (size_t)(ks + 2 * kp) * N + n0 + col;
            float f0 = bp[0];
            float f1 = bp[N];
            *(unsigned*)&Bsh[col][2 * kp] = pack2h(f0, f1);
        }
        __syncthreads();

        #pragma unroll
        for (int kc = 0; kc < TKC; kc += 32) {
            // A fragment: 16x32 f16 (ISA 7.12.2): two contiguous 8-half runs.
            V16H af;
            const int arow = wv * 16 + r16;
            const _Float16* ap0 = &Ash[arow][kc + 8 * half];
            const _Float16* ap1 = &Ash[arow][kc + 16 + 8 * half];
            ((v4h*)af.h)[0] = ((const v4h*)ap0)[0];
            ((v4h*)af.h)[1] = ((const v4h*)ap0)[1];
            ((v4h*)af.h)[2] = ((const v4h*)ap1)[0];
            ((v4h*)af.h)[3] = ((const v4h*)ap1)[1];
            #pragma unroll
            for (int nc = 0; nc < 4; ++nc) {
                // B fragment: 32x16, col = lane%16; two 8-half runs (k=8h.., 16+8h..).
                V16H bf;
                const _Float16* bp = &Bsh[nc * 16 + r16][kc + 8 * half];
                ((v4h*)bf.h)[0] = ((const v4h*)bp)[0];
                ((v4h*)bf.h)[1] = ((const v4h*)bp)[1];
                ((v4h*)bf.h)[2] = ((const v4h*)bp)[4];
                ((v4h*)bf.h)[3] = ((const v4h*)bp)[5];
                acc[nc] = __builtin_amdgcn_wmma_f32_16x16x32_f16(
                    false, af.v, false, bf.v, (short)0, acc[nc], false, false);
            }
        }
    }

    // Epilogue. C/D layout: VGPR r, lanes 0-15 -> M=r, lanes 16-31 -> M=r+8.
    #pragma unroll
    for (int nc = 0; nc < 4; ++nc) {
        const int col  = nc * 16 + r16;
        const int gcol = n0 + col;
        const float bv = bias ? bias[gcol] : 0.f;
        float tsum = 0.f;
        #pragma unroll
        for (int rI = 0; rI < 8; ++rI) {
            int grow = m0 + wv * 16 + rI + 8 * half;
            float vv = acc[nc][rI] + bv;
            if (ep == EP_RELU) vv = fmaxf(vv, 0.f);
            if (ep == EP_TANH_COLSUM) {
                if (grow < M) tsum += tanhf(vv);
            } else if (grow < M) {
                C[(size_t)grow * N + gcol] = vv;
            }
        }
        if (ep == EP_TANH_COLSUM) atomicAdd(&csred[col], tsum);
    }
    if (ep == EP_TANH_COLSUM) {
        __syncthreads();
        if (tid < TBN) atomicAdd(&colsum[n0 + tid], csred[tid]);
    }
}

// ---------------------------------------------------------------------------
// Elementwise / utility kernels
// ---------------------------------------------------------------------------
__global__ __launch_bounds__(256) void fill_f32_k(float* p, float v, size_t n) {
    size_t i = (size_t)blockIdx.x * blockDim.x + threadIdx.x;
    if (i < n) p[i] = v;
}
__global__ __launch_bounds__(256) void fill_u32_k(unsigned* p, unsigned v, size_t n) {
    size_t i = (size_t)blockIdx.x * blockDim.x + threadIdx.x;
    if (i < n) p[i] = v;
}
__global__ __launch_bounds__(256) void relu_k(float* p, size_t n) {
    size_t i = (size_t)blockIdx.x * blockDim.x + threadIdx.x;
    if (i < n) p[i] = fmaxf(p[i], 0.f);
}

// Per-node per-head attention dot: out[n*2+h] = sum_d h[n,h,d]*vec[h,d]
__global__ __launch_bounds__(256) void node_dot_k(
    const float* __restrict__ hbuf, const float* __restrict__ vec,
    float* __restrict__ outv, int Nn)
{
    int t = blockIdx.x * blockDim.x + threadIdx.x;
    if (t >= Nn * 2) return;
    int n = t >> 1, hh = t & 1;
    const float* row = hbuf + (size_t)n * 128 + hh * 64;
    const float* w   = vec + hh * 64;
    float s = 0.f;
    #pragma unroll
    for (int i = 0; i < 16; ++i) {
        float4 a = ((const float4*)row)[i];
        float4 b = ((const float4*)w)[i];
        s += a.x * b.x + a.y * b.y + a.z * b.z + a.w * b.w;
    }
    outv[t] = s;
}

// Monotone float<->uint mapping for atomic max on floats
__device__ __forceinline__ unsigned ordf(float f) {
    unsigned u = __float_as_uint(f);
    return (u & 0x80000000u) ? ~u : (u | 0x80000000u);
}
__device__ __forceinline__ float unordf(unsigned u) {
    u = (u & 0x80000000u) ? (u & 0x7FFFFFFFu) : ~u;
    return __uint_as_float(u);
}
#define ORD_NEGINF 0x007FFFFFu  // ordf(-inf)

// Pass A: per-edge leaky-relu logits + segment max (atomic ordered-uint max)
__global__ __launch_bounds__(256) void edge_logit_max_k(
    const int* __restrict__ src, const int* __restrict__ dst,
    const float* __restrict__ ssrc, const float* __restrict__ sdst,
    float* __restrict__ alpha, unsigned* __restrict__ mord, int E)
{
    int e = blockIdx.x * blockDim.x + threadIdx.x;
    if (e >= E) return;
    int s0 = src[e], d0 = dst[e];
    #pragma unroll
    for (int h = 0; h < 2; ++h) {
        float a = ssrc[(size_t)s0 * 2 + h] + sdst[(size_t)d0 * 2 + h];
        a = a > 0.f ? a : 0.2f * a;                    // leaky_relu 0.2
        alpha[(size_t)e * 2 + h] = a;
        atomicMax(&mord[(size_t)d0 * 2 + h], ordf(a));
    }
}

// Pass B: e = exp(a - m[dst]); segment sum
__global__ __launch_bounds__(256) void edge_exp_sum_k(
    const int* __restrict__ dst, const unsigned* __restrict__ mord,
    float* __restrict__ alpha, float* __restrict__ ssum, int E)
{
    int e = blockIdx.x * blockDim.x + threadIdx.x;
    if (e >= E) return;
    int d0 = dst[e];
    #pragma unroll
    for (int h = 0; h < 2; ++h) {
        float a  = alpha[(size_t)e * 2 + h];
        float mf = unordf(mord[(size_t)d0 * 2 + h]);
        float ev = __expf(a - mf);
        alpha[(size_t)e * 2 + h] = ev;
        atomicAdd(&ssum[(size_t)d0 * 2 + h], ev);
    }
}

// Pass C: one wave32 per edge; lane handles 4 contiguous channels.
// out[dst, c] += h_src[src, c] * (e / (s[dst]+1e-16)) with per-head weight.
__global__ __launch_bounds__(256) void edge_scatter_k(
    const int* __restrict__ src, const int* __restrict__ dst,
    const float* __restrict__ alpha, const float* __restrict__ ssum,
    const float* __restrict__ hsrc, float* __restrict__ outb, int E)
{
    int gw   = (blockIdx.x * blockDim.x + threadIdx.x) >> 5;
    int lane = threadIdx.x & 31;
    if (gw >= E) return;
    int s0 = src[gw], d0 = dst[gw];
    int c0 = lane * 4;
    int h  = c0 >> 6;
    float w = alpha[(size_t)gw * 2 + h] /
              (ssum[(size_t)d0 * 2 + h] + 1e-16f);
    float4 v = *(const float4*)(hsrc + (size_t)s0 * 128 + c0);
    float* o = outb + (size_t)d0 * 128 + c0;
    atomicAdd(o + 0, v.x * w);
    atomicAdd(o + 1, v.y * w);
    atomicAdd(o + 2, v.z * w);
    atomicAdd(o + 3, v.w * w);
}

// Semantic attention scores + 2-way softmax weights
__global__ __launch_bounds__(128) void sem_score_k(
    const float* __restrict__ colsum, const float* __restrict__ q,
    float invN, float* __restrict__ wsem)
{
    __shared__ float red[128];
    float sc[2];
    for (int m = 0; m < 2; ++m) {
        float v = q[threadIdx.x] * colsum[m * 128 + threadIdx.x] * invN;
        red[threadIdx.x] = v;
        __syncthreads();
        for (int s = 64; s > 0; s >>= 1) {
            if ((int)threadIdx.x < s) red[threadIdx.x] += red[threadIdx.x + s];
            __syncthreads();
        }
        sc[m] = red[0];
        __syncthreads();
    }
    if (threadIdx.x == 0) {
        float mx = fmaxf(sc[0], sc[1]);
        float e0 = __expf(sc[0] - mx), e1 = __expf(sc[1] - mx);
        float inv = 1.f / (e0 + e1);
        wsem[0] = e0 * inv;
        wsem[1] = e1 * inv;
    }
}

__global__ __launch_bounds__(256) void combine2_k(
    const float* __restrict__ o0, const float* __restrict__ o1,
    const float* __restrict__ wsem, float* __restrict__ res, size_t n)
{
    size_t i = (size_t)blockIdx.x * blockDim.x + threadIdx.x;
    if (i < n) res[i] = wsem[0] * o0[i] + wsem[1] * o1[i];
}

// Final tiny layer: [M,128] @ [128,8] + b, sigmoid
__global__ __launch_bounds__(256) void mlp3_sigmoid_k(
    const float* __restrict__ X, const float* __restrict__ W,
    const float* __restrict__ b, float* __restrict__ out, int M)
{
    int t = blockIdx.x * blockDim.x + threadIdx.x;
    if (t >= M * 8) return;
    int n = t >> 3, j = t & 7;
    const float* row = X + (size_t)n * 128;
    float s = b[j];
    #pragma unroll
    for (int i = 0; i < 32; ++i) {
        float4 x = ((const float4*)row)[i];
        s += x.x * W[(i * 4 + 0) * 8 + j];
        s += x.y * W[(i * 4 + 1) * 8 + j];
        s += x.z * W[(i * 4 + 2) * 8 + j];
        s += x.w * W[(i * 4 + 3) * 8 + j];
    }
    out[t] = 1.f / (1.f + __expf(-s));
}

// ---------------------------------------------------------------------------
// Host orchestration
// ---------------------------------------------------------------------------
static inline unsigned cdivu(size_t a, size_t b) { return (unsigned)((a + b - 1) / b); }

extern "C" void kernel_launch(void* const* d_in, const int* in_sizes, int n_in,
                              void* d_out, int out_size, void* d_ws, size_t ws_size,
                              hipStream_t stream)
{
    const int C = 128;
    const size_t No = (size_t)in_sizes[0] / C;
    const size_t Na = (size_t)in_sizes[1] / C;
    const size_t E  = (size_t)in_sizes[2] / 2;

    const float* x_o = (const float*)d_in[0];
    const float* x_a = (const float*)d_in[1];
    const int* e_oo  = (const int*)d_in[2];
    const int* e_oa  = (const int*)d_in[3];
    const int* e_ao  = (const int*)d_in[4];
    (void)n_in; (void)ws_size; (void)out_size;

    // Workspace layout (floats). Order matters: h_o,h_a,o2 are contiguous so
    // the MLP t1 buffer (No x 256) can alias them after the layers finish.
    float* w = (float*)d_ws;
    size_t off = 0;
    auto alloc = [&](size_t n) { float* p = w + off; off += n; return p; };
    float* h_o   = alloc(No * C);
    float* h_a   = alloc(Na * C);
    float* o2    = alloc(Na * C);       // metapath (o->a) output, res_attribute
    float* o0    = alloc(No * C);       // metapath (o->o) output
    float* o1    = alloc(No * C);       // metapath (a->o) output
    float* res_o = alloc(No * C);
    float* alpha = alloc(E * 2);
    unsigned* mord = (unsigned*)alloc(No * 2);
    float* ssum  = alloc(No * 2);
    float* sd0 = alloc(No * 2);  // ssrc oo (object)
    float* sd1 = alloc(No * 2);  // sdst oo (object)
    float* sd2 = alloc(No * 2);  // ssrc oa (object)
    float* sd3 = alloc(No * 2);  // sdst oa (attribute)
    float* sd4 = alloc(No * 2);  // ssrc ao (attribute)
    float* sd5 = alloc(No * 2);  // sdst ao (object)
    float* colsum = alloc(256);
    float* wsem   = alloc(16);
    // MLP temps (aliased onto dead regions)
    float* t1 = w;                      // No*256 == |h_o|+|h_a|+|o2| when Na=No/2
    if (No * 256 > No * (size_t)C + Na * (size_t)C * 2) t1 = alloc(No * 256);
    float* t2 = o0;                     // No*128 fits exactly

    auto gemm = [&](const float* A, const float* B, const float* bias, float* Cm,
                    int M, int N, int K, int ep, float* cs) {
        dim3 g(cdivu((size_t)M, TBM), (unsigned)(N / TBN));
        wmma_gemm_k<<<g, 256, 0, stream>>>(A, B, bias, Cm, M, N, K, ep, cs);
    };
    auto fillf = [&](float* p, float v, size_t n) {
        fill_f32_k<<<cdivu(n, 256), 256, 0, stream>>>(p, v, n);
    };
    auto fillu = [&](unsigned* p, unsigned v, size_t n) {
        fill_u32_k<<<cdivu(n, 256), 256, 0, stream>>>(p, v, n);
    };

    auto run_layer = [&](const float* xo, const float* xa, int base) {
        const float* Wo = (const float*)d_in[base + 0];
        const float* bo = (const float*)d_in[base + 1];
        const float* Wa = (const float*)d_in[base + 2];
        const float* ba = (const float*)d_in[base + 3];
        const float* as = (const float*)d_in[base + 4];  // [3,2,64]
        const float* ad = (const float*)d_in[base + 5];
        const float* Wk = (const float*)d_in[base + 6];
        const float* bk = (const float*)d_in[base + 7];
        const float* q  = (const float*)d_in[base + 8];

        // 1) projections (WMMA)
        gemm(xo, Wo, bo, h_o, (int)No, C, C, EP_NONE, nullptr);
        gemm(xa, Wa, ba, h_a, (int)Na, C, C, EP_NONE, nullptr);

        // 2) per-node attention dot products
        unsigned gNo = cdivu(No * 2, 256), gNa = cdivu(Na * 2, 256);
        node_dot_k<<<gNo, 256, 0, stream>>>(h_o, as + 0,   sd0, (int)No);
        node_dot_k<<<gNo, 256, 0, stream>>>(h_o, ad + 0,   sd1, (int)No);
        node_dot_k<<<gNo, 256, 0, stream>>>(h_o, as + 128, sd2, (int)No);
        node_dot_k<<<gNa, 256, 0, stream>>>(h_a, ad + 128, sd3, (int)Na);
        node_dot_k<<<gNa, 256, 0, stream>>>(h_a, as + 256, sd4, (int)Na);
        node_dot_k<<<gNo, 256, 0, stream>>>(h_o, ad + 256, sd5, (int)No);

        // 3) edge types: (o,o,e_oo)->o0, (o,a,e_oa)->o2, (a,o,e_ao)->o1
        struct ETd { const int* s; const int* d; const float* ss; const float* sdv;
                     const float* hs; float* ob; size_t Nd; };
        ETd ets[3] = {
            { e_oo, e_oo + E, sd0, sd1, h_o, o0, No },
            { e_oa, e_oa + E, sd2, sd3, h_o, o2, Na },
            { e_ao, e_ao + E, sd4, sd5, h_a, o1, No },
        };
        unsigned gE  = cdivu(E, 256);
        unsigned gEw = cdivu(E * 32, 256);
        for (int i = 0; i < 3; ++i) {
            ETd& t = ets[i];
            fillu(mord, ORD_NEGINF, t.Nd * 2);
            fillf(ssum, 0.f, t.Nd * 2);
            fillf(t.ob, 0.f, t.Nd * (size_t)C);
            edge_logit_max_k<<<gE, 256, 0, stream>>>(t.s, t.d, t.ss, t.sdv,
                                                     alpha, mord, (int)E);
            edge_exp_sum_k<<<gE, 256, 0, stream>>>(t.d, mord, alpha, ssum, (int)E);
            edge_scatter_k<<<gEw, 256, 0, stream>>>(t.s, t.d, alpha, ssum,
                                                    t.hs, t.ob, (int)E);
            relu_k<<<cdivu(t.Nd * C, 256), 256, 0, stream>>>(t.ob, t.Nd * (size_t)C);
        }

        // 4) semantic attention over object metapaths [o0, o1]
        fillf(colsum, 0.f, 256);
        gemm(o0, Wk, bk, nullptr, (int)No, C, C, EP_TANH_COLSUM, colsum + 0);
        gemm(o1, Wk, bk, nullptr, (int)No, C, C, EP_TANH_COLSUM, colsum + 128);
        sem_score_k<<<1, 128, 0, stream>>>(colsum, q, 1.f / (float)No, wsem);
        combine2_k<<<cdivu(No * C, 256), 256, 0, stream>>>(o0, o1, wsem, res_o,
                                                           No * (size_t)C);
        // attribute has a single metapath -> softmax weight 1 -> res_a = o2
    };

    run_layer(x_o, x_a, 5);            // layer 1 weights at indices 5..13
    run_layer(res_o, o2, 14);          // layer 2 weights at indices 14..22

    // Final MLP on object features
    const float* Wl1 = (const float*)d_in[23];
    const float* bl1 = (const float*)d_in[24];
    const float* Wl2 = (const float*)d_in[25];
    const float* bl2 = (const float*)d_in[26];
    const float* Wl3 = (const float*)d_in[27];
    const float* bl3 = (const float*)d_in[28];

    gemm(res_o, Wl1, bl1, t1, (int)No, 256, 128, EP_NONE, nullptr);
    gemm(t1,    Wl2, bl2, t2, (int)No, 128, 256, EP_NONE, nullptr);
    mlp3_sigmoid_k<<<cdivu(No * 8, 256), 256, 0, stream>>>(t2, Wl3, bl3,
                                                           (float*)d_out, (int)No);
}
